// BasicBlock_27805618274719
// MI455X (gfx1250) — 8-bit matrix operand order repaired and checked by reading, compiled, not run on hardware
//
#include <hip/hip_runtime.h>
#include <cstdint>
#include <cstddef>


typedef __attribute__((ext_vector_type(8))) int v8i;
typedef __attribute__((ext_vector_type(4))) int v4i;
typedef __attribute__((ext_vector_type(2))) int v2i;

#define NB   64
#define CH   256
#define HH   28
#define WW   28
#define HWSZ (HH * WW)
#define NPIX (NB * HWSZ)
#define EPSV 1e-5f

__global__ void pack_x_kernel(const float* __restrict__ x, signed char* __restrict__ sp)
{
    const int d   = blockIdx.x * blockDim.x + threadIdx.x;
    const int pix = d >> 6;
    const int c4  = (d & 63) << 2;
    const int n   = pix / HWSZ;
    const int hw  = pix - n * HWSZ;
    unsigned r = 0;
#pragma unroll
    for (int j = 0; j < 4; ++j) {
        const float v = x[((size_t)(n * CH + c4 + j)) * HWSZ + hw];
        r |= ((v >= 0.f) ? 0x01u : 0xFFu) << (8 * j);
    }
    ((unsigned*)sp)[d] = r;
}

__global__ void pack_w_kernel(const float* __restrict__ w, signed char* __restrict__ wp)
{
    const int d   = blockIdx.x * blockDim.x + threadIdx.x;
    const int t   = d >> 14;
    const int rem = d & 16383;
    const int o   = rem >> 6;
    const int i4  = (rem & 63) << 2;
    unsigned r = 0;
#pragma unroll
    for (int j = 0; j < 4; ++j) {
        const float v = w[(size_t)o * (CH * 9) + (size_t)(i4 + j) * 9 + t];
        r |= ((v >= 0.f) ? 0x01u : 0xFFu) << (8 * j);
    }
    ((unsigned*)wp)[d] = r;
}

__global__ __launch_bounds__(256)
void bconv_wmma_kernel(const signed char* __restrict__ sp,
                       const signed char* __restrict__ wp,
                       float* __restrict__ convout,
                       float* __restrict__ sum, float* __restrict__ sq)
{
    __shared__ signed char lA[9 * 16 * CH];

    const int tid = threadIdx.x;
    const int pb  = blockIdx.x << 4;

    {
        const int prow = tid >> 4;
        const int c16  = (tid & 15) << 4;
        const int p  = pb + prow;
        const int n  = p / HWSZ;
        const int hw = p - n * HWSZ;
        const int h  = hw / WW;
        const int wc = hw - h * WW;
#pragma unroll
        for (int t = 0; t < 9; ++t) {
            const int dy = t / 3 - 1;
            const int dx = t - (t / 3) * 3 - 1;
            const int hs = h + dy;
            const int wsc = wc + dx;
            const bool valid = (hs >= 0) & (hs < HH) & (wsc >= 0) & (wsc < WW);
            v4i val = {0, 0, 0, 0};
            if (valid)
                val = *(const v4i*)(sp + ((size_t)(n * HWSZ + hs * WW + wsc)) * CH + c16);
            *(v4i*)(&lA[(t * 16 + prow) * CH + c16]) = val;
        }
    }
    __syncthreads();

    const int lane = tid & 31;
    const int wave = tid >> 5;
    const int half = lane >> 4;
    const int o0   = (wave << 5) + (lane & 15);
    const int aoff = half * 16;

    v8i acc0 = {0, 0, 0, 0, 0, 0, 0, 0};
    v8i acc1 = {0, 0, 0, 0, 0, 0, 0, 0};

    for (int t = 0; t < 9; ++t) {
        const signed char* la  = &lA[(t * 16 + (lane & 15)) * CH] + aoff;
        const signed char* b0p = wp + (size_t)t * (CH * CH) + (size_t)o0 * CH + half * 16;
        const signed char* b1p = b0p + 16 * CH;
#pragma unroll
        for (int kc = 0; kc < 4; ++kc) {
            const int kb = kc << 6;
            v2i a01 = *(const v2i*)(la + kb);
            v2i a23 = *(const v2i*)(la + kb + 8);
            v2i a45 = *(const v2i*)(la + kb + 32);
            v2i a67 = *(const v2i*)(la + kb + 40);
            v8i A = {a01.x, a01.y, a23.x, a23.y, a45.x, a45.y, a67.x, a67.y};
            v4i c0 = *(const v4i*)(b0p + kb);
            v4i c1 = *(const v4i*)(b0p + kb + 32);
            v8i B0 = {c0.x, c0.y, c0.z, c0.w, c1.x, c1.y, c1.z, c1.w};
            v4i d0 = *(const v4i*)(b1p + kb);
            v4i d1 = *(const v4i*)(b1p + kb + 32);
            v8i B1 = {d0.x, d0.y, d0.z, d0.w, d1.x, d1.y, d1.z, d1.w};
            acc0 = __builtin_amdgcn_wmma_i32_16x16x64_iu8(true, A, true, B0, acc0,
                                                          false, false);
            acc1 = __builtin_amdgcn_wmma_i32_16x16x64_iu8(true, A, true, B1, acc1,
                                                          false, false);
        }
    }

    float ps0 = 0.f, pq0 = 0.f, ps1 = 0.f, pq1 = 0.f;
#pragma unroll
    for (int r = 0; r < 8; ++r) {
        const int m = half * 8 + r;
        const size_t row = ((size_t)(pb + m)) * CH;
        const float v0 = (float)acc0[r];
        const float v1 = (float)acc1[r];
        convout[row + o0]      = v0;
        convout[row + o0 + 16] = v1;
        ps0 += v0; pq0 += v0 * v0;
        ps1 += v1; pq1 += v1 * v1;
    }
    atomicAdd(&sum[o0], ps0);      atomicAdd(&sq[o0], pq0);
    atomicAdd(&sum[o0 + 16], ps1); atomicAdd(&sq[o0 + 16], pq1);
}

__global__ void stats_kernel(const float* __restrict__ sum, const float* __restrict__ sq,
                             float* __restrict__ mu, float* __restrict__ istd)
{
    const int c = threadIdx.x;
    const float inv = 1.0f / (float)NPIX;
    const float m = sum[c] * inv;
    const float v = sq[c] * inv - m * m;
    mu[c]   = m;
    istd[c] = rsqrtf(v + EPSV);
}

__global__ void bn_res_htanh_kernel(const float* __restrict__ conv,
                                    const float* __restrict__ resid,
                                    const float* __restrict__ gamma,
                                    const float* __restrict__ beta,
                                    const float* __restrict__ mu,
                                    const float* __restrict__ istd,
                                    float* __restrict__ out,
                                    signed char* __restrict__ packOut,
                                    int doPack)
{
    const int idx = blockIdx.x * blockDim.x + threadIdx.x;
    const int pix = idx >> 8;
    const int c   = idx & 255;
    const int n   = pix / HWSZ;
    const int hw  = pix - n * HWSZ;
    const size_t xi = ((size_t)(n * CH + c)) * HWSZ + hw;
    float y = gamma[c] * (conv[idx] - mu[c]) * istd[c] + beta[c] + resid[xi];
    y = fminf(1.f, fmaxf(-1.f, y));
    out[xi] = y;
    if (doPack)
        packOut[idx] = (y >= 0.f) ? (signed char)1 : (signed char)-1;
}

extern "C" void kernel_launch(void* const* d_in, const int* in_sizes, int n_in,
                              void* d_out, int out_size, void* d_ws, size_t ws_size,
                              hipStream_t stream)
{
    (void)in_sizes; (void)n_in; (void)out_size; (void)ws_size;
    const float* x  = (const float*)d_in[0];
    const float* w1 = (const float*)d_in[1];
    const float* g1 = (const float*)d_in[2];
    const float* b1 = (const float*)d_in[3];
    const float* w2 = (const float*)d_in[4];
    const float* g2 = (const float*)d_in[5];
    const float* b2 = (const float*)d_in[6];
    float* out = (float*)d_out;

    char* ws = (char*)d_ws;
    signed char* sp  = (signed char*)(ws);
    signed char* wp1 = (signed char*)(ws + 12845056);
    signed char* wp2 = (signed char*)(ws + 13434880);
    float* conv      = (float*)(ws + 14024704);
    float* out1      = (float*)(ws + 65404928);
    float* stats     = (float*)(ws + 116785152);
    float* sum1 = stats;        float* sq1 = stats + 256;
    float* mu1  = stats + 512;  float* is1 = stats + 768;
    float* sum2 = stats + 1024; float* sq2 = stats + 1280;
    float* mu2  = stats + 1536; float* is2 = stats + 1792;

    hipMemsetAsync(stats, 0, 8 * 256 * sizeof(float), stream);

    pack_x_kernel<<<12544, 256, 0, stream>>>(x, sp);
    pack_w_kernel<<<576, 256, 0, stream>>>(w1, wp1);
    pack_w_kernel<<<576, 256, 0, stream>>>(w2, wp2);

    bconv_wmma_kernel<<<3136, 256, 0, stream>>>(sp, wp1, conv, sum1, sq1);
    stats_kernel<<<1, 256, 0, stream>>>(sum1, sq1, mu1, is1);
    bn_res_htanh_kernel<<<50176, 256, 0, stream>>>(conv, x, g1, b1, mu1, is1,
                                                   out1, sp, 1);

    bconv_wmma_kernel<<<3136, 256, 0, stream>>>(sp, wp2, conv, sum2, sq2);
    stats_kernel<<<1, 256, 0, stream>>>(sum2, sq2, mu2, is2);
    bn_res_htanh_kernel<<<50176, 256, 0, stream>>>(conv, out1, g2, b2, mu2, is2,
                                                   out, nullptr, 0);
}
